// SequenceOneChannel_9990093930773
// MI455X (gfx1250) — matrix operand order repaired and checked by reading, not compiled, not run on hardware
//
#include <hip/hip_runtime.h>

typedef __attribute__((ext_vector_type(16))) _Float16 v16h;
typedef __attribute__((ext_vector_type(8)))  float    v8f;

#define NH      15      // hidden size
#define BATCH   4096
#define TILE_B  16      // batch rows per wave (WMMA M)
#define ROWW    24      // LDS row width in halves (48B = 3*16B, keeps b128 align)

// gfx1250 native v_tanh_f32 (TRANS op)
#if __has_builtin(__builtin_amdgcn_tanhf)
__device__ __forceinline__ float tanh_f(float x) {
    return __builtin_amdgcn_tanhf(x);
}
#else
__device__ __forceinline__ float tanh_f(float x) {
    return 1.0f - 2.0f * __builtin_amdgcn_rcpf(1.0f + __expf(2.0f * x));
}
#endif
__device__ __forceinline__ float sigf(float x) {
    return __builtin_fmaf(0.5f, tanh_f(0.5f * x), 0.5f);   // mul + v_tanh + fma
}

__global__ void __launch_bounds__(32)
lstm_seq_wmma_kernel(const float* __restrict__ input,   // [B, T]
                     const float* __restrict__ W_ih1,   // [60, 1]
                     const float* __restrict__ W_hh1,   // [60, 15]
                     const float* __restrict__ b_ih1,   // [60]
                     const float* __restrict__ b_hh1,   // [60]
                     const float* __restrict__ W_ih2,   // [60, 15]
                     const float* __restrict__ W_hh2,   // [60, 15]
                     const float* __restrict__ b_ih2,   // [60]
                     const float* __restrict__ b_hh2,   // [60]
                     const float* __restrict__ W_lin,   // [1, 15]
                     const float* __restrict__ b_lin,   // [1]
                     float* __restrict__ out,           // [B, T+future]
                     int T, int TT)
{
    // One wave (32 lanes) per block; each wave owns 16 batch rows.
    // Rows are 24 halves: cols 0..14 = h, col 15 = x-slot (K=15), 16..23 scrap.
    __shared__ __align__(16) _Float16 lds_h1[TILE_B][ROWW];
    __shared__ __align__(16) _Float16 lds_h2[TILE_B][ROWW];

    const int lane = threadIdx.x & 31;
    const int half = lane >> 4;       // 0: lanes 0-15, 1: lanes 16-31
    const int j    = lane & 15;       // A-row index AND C/D column index
    const int b0   = blockIdx.x * TILE_B;
    const int stCol = (j < NH) ? j : 16;   // j==15 lane writes scrap column

    // ---- init hidden state (and scrap) -------------------------------
    if (lane < 16) {
        #pragma unroll
        for (int k = 0; k < ROWW; ++k) {
            lds_h1[lane][k] = (_Float16)0.0f;
            lds_h2[lane][k] = (_Float16)0.0f;
        }
    }

    // ---- resident weight B-tiles (loaded once) -----------------------
    // B element e -> (K = e<8 ? 8*half+e : 16+8*half+e-8, N = j). Gate-major tiles so each D
    // tile is one whole gate with matching (row, j) slots across tiles.
    // Layer-1 B carries W_ih1 in the K==15 row (x_t rides A's pad slot).
    v16h Bhh1[4], Bih2[4], Bhh2[4], Blin;
    #pragma unroll
    for (int tg = 0; tg < 4; ++tg) {
        v16h b1, b2, b3;
        #pragma unroll
        for (int e = 0; e < 16; ++e) {
            const int K = (e < 8) ? (8 * half + e) : (16 + 8 * half + (e - 8));
            float v1 = 0.0f, v2 = 0.0f, v3 = 0.0f;
            if (j < NH) {
                const int row = tg * NH + j;
                if (K < NH) {
                    v1 = W_hh1[row * NH + K];
                    v2 = W_ih2[row * NH + K];
                    v3 = W_hh2[row * NH + K];
                } else if (K == NH) {
                    v1 = W_ih1[row];
                }
            }
            b1[e] = (_Float16)v1; b2[e] = (_Float16)v2; b3[e] = (_Float16)v3;
        }
        Bhh1[tg] = b1; Bih2[tg] = b2; Bhh2[tg] = b3;
    }
    // Linear head as a B-tile: column N=0 = w_lin, all other columns 0.
    #pragma unroll
    for (int e = 0; e < 16; ++e) {
        const int K = (e < 8) ? (8 * half + e) : (16 + 8 * half + (e - 8));
        Blin[e] = (_Float16)((j == 0 && K < NH) ? W_lin[K] : 0.0f);
    }

    // ---- per-lane constants ------------------------------------------
    float bias1[4], bias2[4];
    #pragma unroll
    for (int tg = 0; tg < 4; ++tg) {
        const int row = tg * NH + j;
        bias1[tg] = (j < NH) ? (b_ih1[row] + b_hh1[row]) : 0.0f;
        bias2[tg] = (j < NH) ? (b_ih2[row] + b_hh2[row]) : 0.0f;
    }
    const float bl = b_lin[0];

    // per-row output pointers (used by lanes with j==0)
    float* rowp[8];
    #pragma unroll
    for (int r = 0; r < 8; ++r)
        rowp[r] = out + (size_t)(b0 + r + 8 * half) * TT;

    float c1v[8], c2v[8];
    #pragma unroll
    for (int r = 0; r < 8; ++r) { c1v[r] = 0.0f; c2v[r] = 0.0f; }

    // A-operand build: row m=j, elements e<8 cover K = 8*half + e (K 0..15);
    // e>=8 map to K>=16 which is always zero padding.
    auto build_a = [&](const _Float16 (*lds)[ROWW]) -> v16h {
        v16h a;
        const _Float16* row = &lds[j][8 * half];   // 16B aligned -> ds_load_b128
        #pragma unroll
        for (int e = 0; e < 8; ++e) a[e] = row[e];
        #pragma unroll
        for (int e = 8; e < 16; ++e) a[e] = (_Float16)0.0f;
        return a;
    };

    // all 32 lanes store unconditionally (j==15 -> scrap col): no exec toggling
    auto gates_update = [&](const v8f* d, float* cv, _Float16 (*lds)[ROWW]) {
        _Float16* stp = &lds[8 * half][stCol];
        #pragma unroll
        for (int r = 0; r < 8; ++r) {
            const float iv = sigf(d[0][r]);
            const float fv = sigf(d[1][r]);
            const float gv = tanh_f(d[2][r]);
            const float ov = sigf(d[3][r]);
            const float cn = fv * cv[r] + iv * gv;
            cv[r] = cn;
            const float hv = ov * tanh_f(cn);
            stp[r * ROWW] = (_Float16)hv;          // immediate-offset ds_store_b16
        }
    };

    auto layer1 = [&](v8f* d) {
        const v16h a1 = build_a(lds_h1);
        #pragma unroll
        for (int tg = 0; tg < 4; ++tg) {
            v8f c;
            #pragma unroll
            for (int r = 0; r < 8; ++r) c[r] = bias1[tg];
            d[tg] = __builtin_amdgcn_wmma_f32_16x16x32_f16(
                        false, a1, false, Bhh1[tg], (short)0, c, false, false);
        }
    };

    auto layer2 = [&](v8f* d, const v16h a2b) {
        const v16h a2a = build_a(lds_h1);   // h1 new; its K=15 x-slot hits Bih2 zeros
        #pragma unroll
        for (int tg = 0; tg < 4; ++tg) {
            v8f c;
            #pragma unroll
            for (int r = 0; r < 8; ++r) c[r] = bias2[tg];
            c = __builtin_amdgcn_wmma_f32_16x16x32_f16(
                    false, a2a, false, Bih2[tg], (short)0, c, false, false);
            c = __builtin_amdgcn_wmma_f32_16x16x32_f16(
                    false, a2b, false, Bhh2[tg], (short)0, c, false, false);
            d[tg] = c;
        }
    };

    auto head = [&](const v16h a2b) -> v8f {                // out = h2 @ w_lin + b
        v8f c;
        #pragma unroll
        for (int r = 0; r < 8; ++r) c[r] = bl;
        return __builtin_amdgcn_wmma_f32_16x16x32_f16(
                   false, a2b, false, Blin, (short)0, c, false, false);
    };

    // lanes l and l+16 index the same input row (j), so duplicate identical
    // LDS writes of x are deterministic: no lane guards in the hot loop.
    const float* xptr = input + (size_t)(b0 + j) * T;
    v8f d[4];

    // ================= main phase: x from input ========================
    if (T > 0) lds_h1[j][15] = (_Float16)xptr[0];          // seed x_0
    for (int t = 0; t < T; ++t) {
        float xnext = 0.0f;
        if (t + 1 < T) xnext = xptr[t + 1];   // prefetch; committed at loop end

        const v16h a2b = build_a(lds_h2);     // h2_{t-1}: feeds head AND layer 2
        const v8f dout = head(a2b);           // out_{t-1}

        layer1(d);                            // issue WMMAs before out-store so
                                              // they cover the head's hazard slots
        if (t > 0 && j == 0) {                // store out_{t-1} (lanes 0 & 16)
            #pragma unroll
            for (int r = 0; r < 8; ++r)
                rowp[r][t - 1] = dout[r];
        }

        gates_update(d, c1v, lds_h1);         // h1_t -> LDS (cols 0..14)
        layer2(d, a2b);                       // reads lds_h1 incl. col15 = x_t
        gates_update(d, c2v, lds_h2);         // h2_t -> LDS

        if (t + 1 < T)                        // commit x_{t+1} AFTER layer2's
            lds_h1[j][15] = (_Float16)xnext;  // ds_load of lds_h1 (in-order DS)
    }

    // ================= future phase: x_t = out_{t-1} ===================
    for (int t = T; t < TT; ++t) {
        const v16h a2b = build_a(lds_h2);
        const v8f dout = head(a2b);           // out_{t-1} == x_t
        if (j == 0) {
            #pragma unroll
            for (int r = 0; r < 8; ++r) {
                const float ov = dout[r];
                rowp[r][t - 1] = ov;
                lds_h1[r + 8 * half][15] = (_Float16)ov;   // feedback x_t
            }
        }
        layer1(d);                            // build_a after DS writes: in-order
        gates_update(d, c1v, lds_h1);
        layer2(d, a2b);
        gates_update(d, c2v, lds_h2);
    }

    // ================= epilogue: out_{TT-1} ============================
    {
        const v16h a2b = build_a(lds_h2);
        const v8f dout = head(a2b);
        if (j == 0) {
            #pragma unroll
            for (int r = 0; r < 8; ++r)
                rowp[r][TT - 1] = dout[r];
        }
    }
}

extern "C" void kernel_launch(void* const* d_in, const int* in_sizes, int n_in,
                              void* d_out, int out_size, void* d_ws, size_t ws_size,
                              hipStream_t stream) {
    const float* input = (const float*)d_in[0];
    const float* W_ih1 = (const float*)d_in[1];
    const float* W_hh1 = (const float*)d_in[2];
    const float* b_ih1 = (const float*)d_in[3];
    const float* b_hh1 = (const float*)d_in[4];
    const float* W_ih2 = (const float*)d_in[5];
    const float* W_hh2 = (const float*)d_in[6];
    const float* b_ih2 = (const float*)d_in[7];
    const float* b_hh2 = (const float*)d_in[8];
    const float* W_lin = (const float*)d_in[9];
    const float* b_lin = (const float*)d_in[10];
    float* out = (float*)d_out;

    const int T  = in_sizes[0] / BATCH;   // 1024
    const int TT = out_size    / BATCH;   // T + future

    dim3 grid(BATCH / TILE_B);            // 256 independent waves
    dim3 block(32);                       // one wave32 per workgroup
    lstm_seq_wmma_kernel<<<grid, block, 0, stream>>>(
        input, W_ih1, W_hh1, b_ih1, b_hh1,
        W_ih2, W_hh2, b_ih2, b_hh2, W_lin, b_lin,
        out, T, TT);
}
